// MambaEncoder_86191403696465
// MI455X (gfx1250) — matrix operand order repaired and checked by reading, not compiled, not run on hardware
//
#include <hip/hip_runtime.h>
#include <hip/hip_bf16.h>
#include <stdint.h>

// ---------------- feature detection: CDNA5 async global->LDS ----------------
#if defined(__has_builtin)
#if __has_builtin(__builtin_amdgcn_global_load_async_to_lds_b128) && \
    __has_builtin(__builtin_amdgcn_s_wait_asynccnt)
#define USE_ASYNC 1
#endif
#endif
#ifndef USE_ASYNC
#define USE_ASYNC 0
#endif

// ---------------- types ----------------
typedef __attribute__((ext_vector_type(16))) __bf16 v16bf;
typedef __attribute__((ext_vector_type(8)))  float  v8f;

#if USE_ASYNC
typedef __attribute__((ext_vector_type(4))) int v4i;
typedef __attribute__((address_space(1))) v4i glb_v4i;   // global int4
typedef __attribute__((address_space(3))) v4i lds_v4i;   // LDS int4
// integer round-trips: always-legal AS conversion.
// generic->LDS: LDS offset lives in the low 32 bits of a generic pointer.
__device__ __forceinline__ lds_v4i* to_lds4(const void* p) {
  return (lds_v4i*)(unsigned int)(unsigned long long)p;
}
// generic->global: numerically identical.
__device__ __forceinline__ glb_v4i* to_glb4(const void* p) {
  return (glb_v4i*)(unsigned long long)p;
}
#endif

#define D_MODEL 1024
#define D_INNER 2048
#define D_STATE 16
#define DT_RANK 64
#define NBATCH  4
#define LSEQ    2048
#define NROW    (NBATCH*LSEQ)   /* 8192 */
#define NCHUNK  16
#define CLEN    (LSEQ/NCHUNK)   /* 128 */

__device__ __forceinline__ unsigned short f2bf(float f) {
  unsigned u = __float_as_uint(f);
  u += 0x7fffu + ((u >> 16) & 1u);
  return (unsigned short)(u >> 16);
}
__device__ __forceinline__ float sigmoid_(float x) { return 1.f / (1.f + __expf(-x)); }

// ---------------- fp32 -> bf16 convert (weights) ----------------
__global__ __launch_bounds__(256)
void convert_kernel(const float* __restrict__ s, unsigned short* __restrict__ d, int n) {
  int i = blockIdx.x * 256 + threadIdx.x;
  if (i < n) d[i] = f2bf(s[i]);
}

// ---------------- embedding gather + layernorm (fused) ----------------
__global__ __launch_bounds__(256)
void embed_ln_kernel(const int* __restrict__ toks, const float* __restrict__ emb,
                     const float* __restrict__ g, const float* __restrict__ be,
                     float* __restrict__ hOut, unsigned short* __restrict__ xnOut) {
  __shared__ float r1[256], r2[256];
  const int row = blockIdx.x, tid = threadIdx.x;
  const float* e = emb + (size_t)toks[row] * D_MODEL;
  float v[4]; float s = 0.f, s2 = 0.f;
#pragma unroll
  for (int i = 0; i < 4; i++) { float x = e[tid + 256 * i]; v[i] = x; s += x; s2 += x * x; }
  r1[tid] = s; r2[tid] = s2; __syncthreads();
  for (int o = 128; o > 0; o >>= 1) {
    if (tid < o) { r1[tid] += r1[tid + o]; r2[tid] += r2[tid + o]; }
    __syncthreads();
  }
  float mean = r1[0] * (1.f / D_MODEL);
  float var  = r2[0] * (1.f / D_MODEL) - mean * mean;
  float rstd = rsqrtf(var + 1e-5f);
#pragma unroll
  for (int i = 0; i < 4; i++) {
    int c = tid + 256 * i;
    float x = v[i];
    hOut[(size_t)row * D_MODEL + c] = x;
    xnOut[(size_t)row * D_MODEL + c] = f2bf((x - mean) * rstd * g[c] + be[c]);
  }
}

// ---------------- layernorm (fp32 in, bf16 out) ----------------
__global__ __launch_bounds__(256)
void ln_bf16_kernel(const float* __restrict__ src, const float* __restrict__ g,
                    const float* __restrict__ be, unsigned short* __restrict__ outBf) {
  __shared__ float r1[256], r2[256];
  const int row = blockIdx.x, tid = threadIdx.x;
  const float* p = src + (size_t)row * D_MODEL;
  float v[4]; float s = 0.f, s2 = 0.f;
#pragma unroll
  for (int i = 0; i < 4; i++) { float x = p[tid + 256 * i]; v[i] = x; s += x; s2 += x * x; }
  r1[tid] = s; r2[tid] = s2; __syncthreads();
  for (int o = 128; o > 0; o >>= 1) {
    if (tid < o) { r1[tid] += r1[tid + o]; r2[tid] += r2[tid + o]; }
    __syncthreads();
  }
  float mean = r1[0] * (1.f / D_MODEL);
  float var  = r2[0] * (1.f / D_MODEL) - mean * mean;
  float rstd = rsqrtf(var + 1e-5f);
#pragma unroll
  for (int i = 0; i < 4; i++) {
    int c = tid + 256 * i;
    outBf[(size_t)row * D_MODEL + c] = f2bf((v[i] - mean) * rstd * g[c] + be[c]);
  }
}

// ---------------- depthwise causal conv (D_CONV=4) + SiLU ----------------
__global__ __launch_bounds__(256)
void conv_silu_kernel(const float* __restrict__ xz, const float* __restrict__ cw,
                      const float* __restrict__ cb,
                      float* __restrict__ uF, unsigned short* __restrict__ uB) {
  int idx = blockIdx.x * 256 + threadIdx.x;
  int d = idx & (D_INNER - 1);
  int row = idx >> 11;
  int l = row & (LSEQ - 1);
  int b = row >> 11;
  float acc = cb[d];
#pragma unroll
  for (int k = 0; k < 4; k++) {
    int ls = l - 3 + k;
    if (ls >= 0)
      acc += cw[d * 4 + k] * xz[(size_t)((b << 11) + ls) * (2 * D_INNER) + d];
  }
  float u = acc * sigmoid_(acc);
  uF[idx] = u;
  uB[idx] = f2bf(u);
}

// ---------------- chunk-parallel selective scan ----------------
// recurrence: h = dA*h + du*B  (linear) => chunk decomposition.
// pass1: per chunk, P = prod(dA), R = h starting from 0
__global__ __launch_bounds__(256)
void scan_pass1(const float* __restrict__ dt, const float* __restrict__ u,
                const float* __restrict__ dbl, const float* __restrict__ A_log,
                float* __restrict__ Pout, float* __restrict__ Rout) {
  int d = blockIdx.x * 256 + threadIdx.x;
  int b = blockIdx.y, c = blockIdx.z;
  float Ar[D_STATE], h[D_STATE], P[D_STATE];
#pragma unroll
  for (int s = 0; s < D_STATE; s++) { Ar[s] = -__expf(A_log[d * D_STATE + s]); h[s] = 0.f; P[s] = 1.f; }
  for (int l = c * CLEN; l < (c + 1) * CLEN; l++) {
    size_t row = ((size_t)b << 11) + l;
    float dtv = dt[row * D_INNER + d];
    float uv  = u[row * D_INNER + d];
    float du  = dtv * uv;
    const float* Br = dbl + row * 96 + DT_RANK;
#pragma unroll
    for (int s = 0; s < D_STATE; s++) {
      float dA = __expf(dtv * Ar[s]);
      h[s] = h[s] * dA + du * Br[s];
      P[s] *= dA;
    }
  }
  size_t base = (((size_t)b * NCHUNK + c) * D_INNER + d) * D_STATE;
#pragma unroll
  for (int s = 0; s < D_STATE; s++) { Pout[base + s] = P[s]; Rout[base + s] = h[s]; }
}

// pass2: stitch chunk boundary states (sequential over only NCHUNK steps)
__global__ __launch_bounds__(256)
void scan_pass2(const float* __restrict__ P, const float* __restrict__ R,
                float* __restrict__ Hin) {
  int idx = blockIdx.x * 256 + threadIdx.x;          // (b, d, s)
  int s = idx & (D_STATE - 1);
  int d = (idx >> 4) & (D_INNER - 1);
  int b = idx >> 15;
  float h = 0.f;
  for (int c = 0; c < NCHUNK; c++) {
    size_t o = (((size_t)b * NCHUNK + c) * D_INNER + d) * D_STATE + s;
    Hin[o] = h;
    h = P[o] * h + R[o];
  }
}

// pass3: replay chunk from true initial state, emit fused (y + u*D)*silu(z) -> bf16
__global__ __launch_bounds__(256)
void scan_pass3(const float* __restrict__ dt, const float* __restrict__ u,
                const float* __restrict__ dbl, const float* __restrict__ xz,
                const float* __restrict__ A_log, const float* __restrict__ Dp,
                const float* __restrict__ Hin, unsigned short* __restrict__ yB) {
  int d = blockIdx.x * 256 + threadIdx.x;
  int b = blockIdx.y, c = blockIdx.z;
  float Ar[D_STATE], h[D_STATE];
  size_t base = (((size_t)b * NCHUNK + c) * D_INNER + d) * D_STATE;
#pragma unroll
  for (int s = 0; s < D_STATE; s++) { Ar[s] = -__expf(A_log[d * D_STATE + s]); h[s] = Hin[base + s]; }
  float Dv = Dp[d];
  for (int l = c * CLEN; l < (c + 1) * CLEN; l++) {
    size_t row = ((size_t)b << 11) + l;
    float dtv = dt[row * D_INNER + d];
    float uv  = u[row * D_INNER + d];
    float du  = dtv * uv;
    const float* Br = dbl + row * 96 + DT_RANK;
    const float* Cr = Br + D_STATE;
    float y = 0.f;
#pragma unroll
    for (int s = 0; s < D_STATE; s++) {
      float dA = __expf(dtv * Ar[s]);
      h[s] = h[s] * dA + du * Br[s];
      y += h[s] * Cr[s];
    }
    float zv = xz[row * (2 * D_INNER) + D_INNER + d];
    float yt = (y + uv * Dv) * (zv * sigmoid_(zv));
    yB[row * D_INNER + d] = f2bf(yt);
  }
}

// ---------------- bf16 WMMA GEMM, 128x128 tile, double-buffered LDS ----------
#define BM 128
#define BN 128
#define BK 32
#define LDSA 40
#define LDSB 40

template<bool RAGGED, int ACT, bool HAS_BIAS, bool HAS_RESID, bool HAS_OUTF, bool HAS_OUTB>
__global__ __launch_bounds__(256)
void gemm_bf16_wmma(const unsigned short* __restrict__ A,
                    const unsigned short* __restrict__ B,
                    int M, int N, int K, int lda, int ldb,
                    const float* __restrict__ bias,
                    const float* __restrict__ resid,
                    float* __restrict__ outF,
                    unsigned short* __restrict__ outB) {
  __shared__ alignas(16) unsigned short sA[2][BM * LDSA];
  __shared__ alignas(16) unsigned short sB[2][BN * LDSB];   // transposed [n][k]
  const int tid = threadIdx.x;
  const int m0 = blockIdx.x * BM, n0 = blockIdx.y * BN;
  const int wid = tid >> 5, lane = tid & 31;
  const int wm = (wid & 1) * 64;            // 2 waves along M, wave tile 64 rows
  const int wn = (wid >> 1) * 32;           // 4 waves along N, wave tile 32 cols
  const int lhalf = lane >> 4, lidx = lane & 15;

  const int arow = tid >> 1, aseg = (tid & 1) * 16;       // A: 2 thr/row, 16 bf16 each
  const int bkp  = (tid >> 4) * 2, bnb = (tid & 15) * 8;  // B: 2 k-rows x 8 cols/thr

  const unsigned short* gA = A + (size_t)(m0 + arow) * lda + aseg;

  alignas(16) uint4 ta0, ta1;
  alignas(16) unsigned short tb0[8], tb1[8];

  auto loadAregs = [&](int k0) {
    const unsigned short* p = gA + k0;
    ta0 = *(const uint4*)p;
    ta1 = *(const uint4*)(p + 8);
  };
  auto storeAlds = [&](int buf) {
    *(uint4*)&sA[buf][arow * LDSA + aseg]     = ta0;
    *(uint4*)&sA[buf][arow * LDSA + aseg + 8] = ta1;
  };
#if USE_ASYNC
  auto asyncA = [&](int k0, int buf) {
    const unsigned short* p = gA + k0;
    const unsigned short* dst = &sA[buf][arow * LDSA + aseg];
    __builtin_amdgcn_global_load_async_to_lds_b128(to_glb4(p),     to_lds4(dst),     0, 0);
    __builtin_amdgcn_global_load_async_to_lds_b128(to_glb4(p + 8), to_lds4(dst + 8), 0, 0);
  };
#endif
  auto loadBregs = [&](int k0) {
    const unsigned short* p0 = B + (size_t)(k0 + bkp) * ldb + (n0 + bnb);
    const unsigned short* p1 = p0 + ldb;
    if (!RAGGED || (n0 + bnb + 8 <= N)) {
      *(uint4*)tb0 = *(const uint4*)p0;
      *(uint4*)tb1 = *(const uint4*)p1;
    } else {
#pragma unroll
      for (int j = 0; j < 8; j++) {
        tb0[j] = (n0 + bnb + j < N) ? p0[j] : (unsigned short)0;
        tb1[j] = (n0 + bnb + j < N) ? p1[j] : (unsigned short)0;
      }
    }
  };
  auto storeBlds = [&](int buf) {
#pragma unroll
    for (int j = 0; j < 8; j++)
      *(unsigned*)&sB[buf][(bnb + j) * LDSB + bkp] =
          (unsigned)tb0[j] | ((unsigned)tb1[j] << 16);
  };

  v8f acc[4][2];
#pragma unroll
  for (int mi = 0; mi < 4; mi++)
#pragma unroll
    for (int ni = 0; ni < 2; ni++)
      acc[mi][ni] = (v8f){0.f, 0.f, 0.f, 0.f, 0.f, 0.f, 0.f, 0.f};

  // prologue: stage k-step 0 into buffer 0
#if USE_ASYNC
  asyncA(0, 0);
#else
  loadAregs(0); storeAlds(0);
#endif
  loadBregs(0); storeBlds(0);
#if USE_ASYNC
  __builtin_amdgcn_s_wait_asynccnt(0);
#endif
  __syncthreads();

  const int nk = K / BK;
  for (int ks = 0; ks < nk; ks++) {
    const int cur = ks & 1;
    const bool notlast = (ks + 1 < nk);
    if (notlast) {
#if USE_ASYNC
      asyncA((ks + 1) * BK, cur ^ 1);   // DMA next A tile while we compute
#else
      loadAregs((ks + 1) * BK);
#endif
      loadBregs((ks + 1) * BK);
    }

    union Frag { v16bf v; unsigned u[8]; } fa[4], fb[2];
#pragma unroll
    for (int mi = 0; mi < 4; mi++) {
      const unsigned short* p = &sA[cur][(wm + mi * 16 + lidx) * LDSA + lhalf * 8];
      fa[mi].u[0] = *(const unsigned*)(p + 0);
      fa[mi].u[1] = *(const unsigned*)(p + 2);
      fa[mi].u[2] = *(const unsigned*)(p + 4);
      fa[mi].u[3] = *(const unsigned*)(p + 6);
      fa[mi].u[4] = *(const unsigned*)(p + 16);
      fa[mi].u[5] = *(const unsigned*)(p + 18);
      fa[mi].u[6] = *(const unsigned*)(p + 20);
      fa[mi].u[7] = *(const unsigned*)(p + 22);
    }
#pragma unroll
    for (int ni = 0; ni < 2; ni++) {
      const unsigned short* p = &sB[cur][(wn + ni * 16 + lidx) * LDSB + lhalf * 8];
#pragma unroll
      for (int i = 0; i < 8; i++) fb[ni].u[i] = *(const unsigned*)(p + 2 * i + (i >= 4 ? 8 : 0));
    }
#pragma unroll
    for (int mi = 0; mi < 4; mi++)
#pragma unroll
      for (int ni = 0; ni < 2; ni++)
        acc[mi][ni] = __builtin_amdgcn_wmma_f32_16x16x32_bf16(
            false, fa[mi].v, false, fb[ni].v, (short)0, acc[mi][ni], false, false);

    if (notlast) {
#if !USE_ASYNC
      storeAlds(cur ^ 1);
#endif
      storeBlds(cur ^ 1);
#if USE_ASYNC
      __builtin_amdgcn_s_wait_asynccnt(0);
#endif
      __syncthreads();
    }
  }

  // epilogue
#pragma unroll
  for (int mi = 0; mi < 4; mi++) {
#pragma unroll
    for (int ni = 0; ni < 2; ni++) {
      int n = n0 + wn + ni * 16 + lidx;
      if (RAGGED && n >= N) continue;
#pragma unroll
      for (int r = 0; r < 8; r++) {
        int m = m0 + wm + mi * 16 + lhalf * 8 + r;
        float v = acc[mi][ni][r];
        if (HAS_BIAS) v += bias[n];
        if (ACT == 1) v = fmaxf(v, 0.f);
        if (ACT == 2) v = (v > 20.f) ? v : log1pf(__expf(v));
        if (HAS_RESID) v += resid[(size_t)m * N + n];
        if (HAS_OUTF) outF[(size_t)m * N + n] = v;
        if (HAS_OUTB) outB[(size_t)m * N + n] = f2bf(v);
      }
    }
  }
}

// ---------------- launch ----------------
extern "C" void kernel_launch(void* const* d_in, const int* in_sizes, int n_in,
                              void* d_out, int out_size, void* d_ws, size_t ws_size,
                              hipStream_t stream) {
  (void)in_sizes; (void)n_in; (void)out_size; (void)ws_size;
  const int*   x        = (const int*)  d_in[0];
  const float* emb      = (const float*)d_in[1];
  const float* ln1_g    = (const float*)d_in[2];
  const float* ln1_b    = (const float*)d_in[3];
  const float* in_proj  = (const float*)d_in[4];
  const float* conv_w   = (const float*)d_in[5];
  const float* conv_b   = (const float*)d_in[6];
  const float* x_proj   = (const float*)d_in[7];
  const float* dt_proj  = (const float*)d_in[8];
  const float* dt_bias  = (const float*)d_in[9];
  const float* A_log    = (const float*)d_in[10];
  const float* Dp       = (const float*)d_in[11];
  const float* out_proj = (const float*)d_in[12];
  const float* ln2_g    = (const float*)d_in[13];
  const float* ln2_b    = (const float*)d_in[14];
  const float* w1       = (const float*)d_in[15];
  const float* b1       = (const float*)d_in[16];
  const float* w2       = (const float*)d_in[17];
  const float* b2       = (const float*)d_in[18];
  float* out = (float*)d_out;

  char* ws = (char*)d_ws;
  size_t off = 0;
  auto alloc = [&](size_t bytes) -> char* {
    char* p = ws + off; off += (bytes + 255) & ~(size_t)255; return p;
  };
  unsigned short* wbIn = (unsigned short*)alloc((size_t)D_MODEL * 2 * D_INNER * 2);
  unsigned short* wbXp = (unsigned short*)alloc((size_t)D_INNER * 96 * 2);
  unsigned short* wbDt = (unsigned short*)alloc((size_t)DT_RANK * D_INNER * 2);
  unsigned short* wbOp = (unsigned short*)alloc((size_t)D_INNER * D_MODEL * 2);
  unsigned short* wbW1 = (unsigned short*)alloc((size_t)D_MODEL * 4 * D_MODEL * 2);
  unsigned short* wbW2 = (unsigned short*)alloc((size_t)4 * D_MODEL * D_MODEL * 2);
  unsigned short* xnBf = (unsigned short*)alloc((size_t)NROW * D_MODEL * 2);
  float*          hRes = (float*)         alloc((size_t)NROW * D_MODEL * 4);
  float*          xzF  = (float*)         alloc((size_t)NROW * 2 * D_INNER * 4);
  float*          uF   = (float*)         alloc((size_t)NROW * D_INNER * 4);
  unsigned short* uBf  = (unsigned short*)alloc((size_t)NROW * D_INNER * 2);
  float*          dblF = (float*)         alloc((size_t)NROW * 96 * 4);
  unsigned short* dblBf= (unsigned short*)alloc((size_t)NROW * 96 * 2);
  float*          dtF  = (float*)         alloc((size_t)NROW * D_INNER * 4);
  unsigned short* yBf  = (unsigned short*)alloc((size_t)NROW * D_INNER * 2);
  float*          h2F  = (float*)         alloc((size_t)NROW * D_MODEL * 4);
  float*          scP  = (float*)         alloc((size_t)NBATCH * NCHUNK * D_INNER * D_STATE * 4);
  float*          scR  = (float*)         alloc((size_t)NBATCH * NCHUNK * D_INNER * D_STATE * 4);
  float*          scH  = (float*)         alloc((size_t)NBATCH * NCHUNK * D_INNER * D_STATE * 4);
  unsigned short* hnBf = xnBf;                  // xn dead after in_proj GEMM
  unsigned short* a1Bf = (unsigned short*)xzF;  // xz dead after scan

  auto cvt = [&](const float* s, unsigned short* d, int n) {
    convert_kernel<<<(n + 255) / 256, 256, 0, stream>>>(s, d, n);
  };

  cvt(in_proj,  wbIn, D_MODEL * 2 * D_INNER);
  cvt(x_proj,   wbXp, D_INNER * 96);
  cvt(dt_proj,  wbDt, DT_RANK * D_INNER);
  cvt(out_proj, wbOp, D_INNER * D_MODEL);
  cvt(w1,       wbW1, D_MODEL * 4 * D_MODEL);
  cvt(w2,       wbW2, 4 * D_MODEL * D_MODEL);

  // 1. embedding + LN1
  embed_ln_kernel<<<NROW, 256, 0, stream>>>(x, emb, ln1_g, ln1_b, hRes, xnBf);
  // 2. xz = xn @ in_proj  (8192x1024 . 1024x4096)
  gemm_bf16_wmma<false, 0, false, false, true, false>
      <<<dim3(NROW / BM, (2 * D_INNER) / BN), 256, 0, stream>>>(
      xnBf, wbIn, NROW, 2 * D_INNER, D_MODEL, D_MODEL, 2 * D_INNER,
      nullptr, nullptr, xzF, nullptr);
  // 3. depthwise conv + SiLU -> u
  conv_silu_kernel<<<(NROW * D_INNER) / 256, 256, 0, stream>>>(xzF, conv_w, conv_b, uF, uBf);
  // 4. dbl = u @ x_proj  (8192x2048 . 2048x96), ragged N
  gemm_bf16_wmma<true, 0, false, false, true, true>
      <<<dim3(NROW / BM, 1), 256, 0, stream>>>(
      uBf, wbXp, NROW, 96, D_INNER, D_INNER, 96, nullptr, nullptr, dblF, dblBf);
  // 5. dt = softplus(dt_in @ dt_proj + b)  (8192x64 . 64x2048)
  gemm_bf16_wmma<false, 2, true, false, true, false>
      <<<dim3(NROW / BM, D_INNER / BN), 256, 0, stream>>>(
      dblBf, wbDt, NROW, D_INNER, DT_RANK, 96, D_INNER, dt_bias, nullptr, dtF, nullptr);
  // 6. chunk-parallel scan
  scan_pass1<<<dim3(D_INNER / 256, NBATCH, NCHUNK), 256, 0, stream>>>(
      dtF, uF, dblF, A_log, scP, scR);
  scan_pass2<<<(NBATCH * D_INNER * D_STATE) / 256, 256, 0, stream>>>(scP, scR, scH);
  scan_pass3<<<dim3(D_INNER / 256, NBATCH, NCHUNK), 256, 0, stream>>>(
      dtF, uF, dblF, xzF, A_log, Dp, scH, yBf);
  // 7. h2 = h + y @ out_proj  (8192x2048 . 2048x1024)
  gemm_bf16_wmma<false, 0, false, true, true, false>
      <<<dim3(NROW / BM, D_MODEL / BN), 256, 0, stream>>>(
      yBf, wbOp, NROW, D_MODEL, D_INNER, D_INNER, D_MODEL, nullptr, hRes, h2F, nullptr);
  // 8. LN2
  ln_bf16_kernel<<<NROW, 256, 0, stream>>>(h2F, ln2_g, ln2_b, hnBf);
  // 9. a1 = relu(hn @ w1 + b1)  (8192x1024 . 1024x4096)
  gemm_bf16_wmma<false, 1, true, false, false, true>
      <<<dim3(NROW / BM, (4 * D_MODEL) / BN), 256, 0, stream>>>(
      hnBf, wbW1, NROW, 4 * D_MODEL, D_MODEL, D_MODEL, 4 * D_MODEL,
      b1, nullptr, nullptr, a1Bf);
  // 10. out = h2 + a1 @ w2 + b2  (8192x4096 . 4096x1024)
  gemm_bf16_wmma<false, 0, true, true, true, false>
      <<<dim3(NROW / BM, D_MODEL / BN), 256, 0, stream>>>(
      a1Bf, wbW2, NROW, D_MODEL, 4 * D_MODEL, 4 * D_MODEL, D_MODEL,
      b2, h2F, out, nullptr);
}
